// DeformableConv2d_65438121722085
// MI455X (gfx1250) — compile-verified, matrix operand order checked by reading, not run on hardware
//
#include <hip/hip_runtime.h>
#include <hip/hip_bf16.h>

typedef __attribute__((ext_vector_type(16))) _Float16 v16h;
typedef __attribute__((ext_vector_type(8)))  _Float16 v8h;
typedef __attribute__((ext_vector_type(8)))  float    v8f;

#define B_   8
#define C_   256
#define H_   64
#define W_   64
#define HP_  66            /* padded (1-pixel zero halo) */
#define WP_  66
#define CO_  256
#define K2_  9
#define KTOT (C_ * K2_)    /* 2304 */
#define NKB  (KTOT / 32)   /* 72 K-blocks of 32 */
#define NPIX (B_ * H_ * W_)    /* 32768 */
#define NTILE_M (NPIX / 16)    /* 2048 */

/* workspace layout (bytes) */
#define XT_OFF   0u
#define XT_BYTES ((size_t)B_ * HP_ * WP_ * C_ * 2)       /* 17.84 MB f16 NHWC+halo */
#define WB_OFF   (XT_OFF + XT_BYTES)
#define WB_BYTES ((size_t)KTOT * CO_ * 2)                /* 1.18 MB swizzled B */
#define WOM_OFF  (WB_OFF + WB_BYTES)
#define WOM_BYTES ((size_t)KTOT * 32 * 2)                /* 147 KB swizzled B */
#define OM_OFF   (WOM_OFF + WOM_BYTES)                   /* 4.19 MB f32 off/mask */

static __device__ __forceinline__ v8f wmma_f16(v16h a, v16h b, v8f c) {
    return __builtin_amdgcn_wmma_f32_16x16x32_f16(false, a, false, b,
                                                  (short)0, c, false, false);
}

static __device__ __forceinline__ size_t xt_idx(int b, int yp, int xp) {
    return ((size_t)(b * HP_ + yp) * WP_ + xp) * C_;
}

/* ---------------- 0a-halo: zero the 1-pixel halo of padded xT ---------------- */
__global__ void k_halo(_Float16* __restrict__ xT) {
    const int b  = blockIdx.x / HP_;
    const int yp = blockIdx.x % HP_;
    const v8h z = {};
    _Float16* row = xT + xt_idx(b, yp, 0);
    if (yp == 0 || yp == HP_ - 1) {
        for (int i = threadIdx.x; i < WP_ * C_ / 8; i += 256)
            ((v8h*)row)[i] = z;
    } else {
        if (threadIdx.x < 32)
            ((v8h*)row)[threadIdx.x] = z;                       /* xp = 0   */
        else if (threadIdx.x < 64)
            ((v8h*)(row + (size_t)(WP_ - 1) * C_))[threadIdx.x - 32] = z; /* xp = 65 */
    }
}

/* ---------------- 0a: NCHW f32 -> NHWC+halo f16 (LDS tiled transpose) -------- */
__global__ void k_xpose(const float* __restrict__ x, _Float16* __restrict__ xT) {
    const int by = blockIdx.x >> 3;            /* (b,y) pair, 0..511 */
    const int c0 = (blockIdx.x & 7) << 5;      /* channel block of 32 */
    const int b  = by >> 6, y = by & 63;
    __shared__ float tile[32][65];
    const int t = threadIdx.x;
    {
        const int xw = t & 63;
        const int cl0 = t >> 6;                /* 0..3 */
#pragma unroll
        for (int i = 0; i < 8; ++i) {
            const int cl = cl0 + i * 4;
            tile[cl][xw] = x[(((size_t)(b * C_ + c0 + cl) * H_ + y) << 6) + xw];
        }
    }
    __syncthreads();
    {
        const int cl = t & 31;
        const int x0 = t >> 5;                 /* 0..7 */
#pragma unroll
        for (int i = 0; i < 8; ++i) {
            const int xx = x0 + i * 8;
            xT[xt_idx(b, y + 1, xx + 1) + c0 + cl] = (_Float16)tile[cl][xx];
        }
    }
}

/* -------- 0b: swizzle reg_w (Co,C,3,3) into per-lane B fragments ------------- */
/* layout: [kblock(72)][ntile(16)][lane(32)][16 halves], K order k = kk*256 + c */
__global__ void k_packB(const float* __restrict__ regw, _Float16* __restrict__ wB) {
    const int t = blockIdx.x * 256 + threadIdx.x;
    if (t >= NKB * 16 * 32 * 16) return;
    const int i    = t & 15;
    const int lane = (t >> 4) & 31;
    const int nt   = (t >> 9) & 15;
    const int kb   = t >> 13;
    const int kbase = (lane >> 4) * 8;
    const int r = (i < 8) ? (kbase + i) : (kbase + 16 + (i - 8));
    const int k = kb * 32 + r;
    const int n = nt * 16 + (lane & 15);
    const int kk = k >> 8, c = k & 255;
    wB[t] = (_Float16)regw[((size_t)(n * C_ + c)) * K2_ + kk];
}

/* -------- 0c: swizzle offset_w(18,C,3,3)+mod_w(9,C,3,3) -> 32-col B ---------- */
__global__ void k_packOM(const float* __restrict__ offw,
                         const float* __restrict__ modw,
                         _Float16* __restrict__ wOM) {
    const int t = blockIdx.x * 256 + threadIdx.x;
    if (t >= NKB * 2 * 32 * 16) return;
    const int i    = t & 15;
    const int lane = (t >> 4) & 31;
    const int nt   = (t >> 9) & 1;
    const int kb   = t >> 10;
    const int kbase = (lane >> 4) * 8;
    const int r = (i < 8) ? (kbase + i) : (kbase + 16 + (i - 8));
    const int k = kb * 32 + r;
    const int n = nt * 16 + (lane & 15);
    const int kk = k >> 8, c = k & 255;
    float v = 0.0f;
    if (n < 18)      v = offw[((size_t)(n * C_ + c)) * K2_ + kk];
    else if (n < 27) v = modw[((size_t)((n - 18) * C_ + c)) * K2_ + kk];
    wOM[t] = (_Float16)v;
}

/* -------- 1: offset/mask conv as implicit GEMM (WMMA), 16 pix x 32 out ------- */
/* halo-padded xT: every im2col load unconditionally valid -> no EXEC branching */
__global__ void k_offmask(const _Float16* __restrict__ xT,
                          const _Float16* __restrict__ wOM,
                          const float* __restrict__ offb,
                          const float* __restrict__ modb,
                          float* __restrict__ om) {
    const int wave = (blockIdx.x << 2) + (threadIdx.x >> 5);   /* 0..2047 */
    const int lane = threadIdx.x & 31;
    const int p0  = wave << 4;
    const int b   = p0 >> 12;
    const int rem = p0 & 4095;
    const int y   = rem >> 6;
    const int x0  = rem & 63;
    const int m   = lane & 15;
    const int kb  = (lane >> 4) << 3;

    v8f acc0 = {}, acc1 = {};
#pragma unroll
    for (int kk = 0; kk < 9; ++kk) {
        const int ky = kk / 3, kx = kk % 3;
        /* padded coords: yp = y+ky-1+1, xp = x0+m+kx-1+1 : always in range */
        const _Float16* rowp = xT + xt_idx(b, y + ky, x0 + m + kx);
#pragma unroll
        for (int cb = 0; cb < 8; ++cb) {
            union { v16h v; v8h h[2]; } af;
            const _Float16* ap = rowp + cb * 32 + kb;
            af.h[0] = *(const v8h*)(ap);
            af.h[1] = *(const v8h*)(ap + 16);
            const int kblock = kk * 8 + cb;
            const v16h bf0 = *(const v16h*)(wOM + (((size_t)kblock * 2 + 0) * 32 + lane) * 16);
            const v16h bf1 = *(const v16h*)(wOM + (((size_t)kblock * 2 + 1) * 32 + lane) * 16);
            acc0 = wmma_f16(af.v, bf0, acc0);
            acc1 = wmma_f16(af.v, bf1, acc1);
        }
    }
    /* epilogue: C/D layout M = r + 8*(lane>=16), N = lane&15 */
    const int nlo = lane & 15;
    const int mhi = (lane >> 4) << 3;
#pragma unroll
    for (int r = 0; r < 8; ++r) {
        const int p = p0 + r + mhi;
        {   /* columns 0..15: offset channels */
            om[(size_t)p * 32 + nlo] = acc0[r] + offb[nlo];
        }
        {   /* columns 16..31: 16,17 offsets; 18..26 mask; rest pad */
            const int j = 16 + nlo;
            float v = acc1[r];
            if (j < 18)      v += offb[j];
            else if (j < 27) { v += modb[j - 18]; v = 2.0f / (1.0f + __expf(-v)); }
            else             v = 0.0f;
            om[(size_t)p * 32 + j] = v;
        }
    }
}

/* -------- 2: deformable GEMM: 64 pixels (one row) x 256 outputs per WG ------- */
/* 8 waves; each wave: 4 M-tiles x 2 N-tiles -> each B fragment feeds 4 WMMAs  */
__global__ void __launch_bounds__(256)
k_deform(const _Float16* __restrict__ xT,
         const _Float16* __restrict__ wB,
         const float* __restrict__ om,
         float* __restrict__ out) {
    const int tid = blockIdx.x;            /* 0..511: one image row each */
    const int p0  = tid << 6;
    const int b   = p0 >> 12;
    const int y   = (p0 >> 6) & 63;        /* x0 == 0 */
    const int t    = threadIdx.x;
    const int lane = t & 31;
    const int wv   = t >> 5;               /* 0..7 */

    __shared__ _Float16 As[64][280];       /* row stride 560B: 16B aligned, bank-spread */
    __shared__ float OM[64][27];

    for (int idx = t; idx < 64 * 27; idx += 256)
        OM[idx / 27][idx % 27] = om[(size_t)(p0 + idx / 27) * 32 + (idx % 27)];
    __syncthreads();

    v8f acc[4][2];
#pragma unroll
    for (int mt = 0; mt < 4; ++mt) {
        acc[mt][0] = (v8f){};
        acc[mt][1] = (v8f){};
    }
    const int nt0 = wv * 2, nt1 = wv * 2 + 1;
    const int m  = lane & 15;
    const int kb = (lane >> 4) << 3;

    for (int kk = 0; kk < 9; ++kk) {
        /* ---- build modulated bilinear sample tile: 64 pixels x 256 ch ---- */
#pragma unroll
        for (int it = 0; it < 4; ++it) {
            const int task = t + it * 256;    /* 0..1023 */
            const int pm = task >> 4;         /* pixel in tile 0..63 */
            const int ch = (task & 15) << 4;  /* 16-channel chunk */
            const float dy  = OM[pm][2 * kk];
            const float dx  = OM[pm][2 * kk + 1];
            const float msk = OM[pm][18 + kk];
            const float yf = (float)(y + kk / 3 - 1) + dy;
            const float xf = (float)(pm + kk % 3 - 1) + dx;
            const float y0f = floorf(yf), x0f = floorf(xf);
            const float wy1 = yf - y0f, wx1 = xf - x0f;
            const float wy0 = 1.0f - wy1, wx0 = 1.0f - wx1;
            const int iy0 = (int)y0f, ix0 = (int)x0f;
            const int iy1 = iy0 + 1,  ix1 = ix0 + 1;
            const bool vy0 = (iy0 >= 0) && (iy0 < H_);
            const bool vy1 = (iy1 >= 0) && (iy1 < H_);
            const bool vx0 = (ix0 >= 0) && (ix0 < W_);
            const bool vx1 = (ix1 >= 0) && (ix1 < W_);
            const int cy0 = min(max(iy0, 0), H_ - 1);
            const int cy1 = min(max(iy1, 0), H_ - 1);
            const int cx0 = min(max(ix0, 0), W_ - 1);
            const int cx1 = min(max(ix1, 0), W_ - 1);
            const float w00 = (vy0 && vx0) ? msk * wy0 * wx0 : 0.0f;
            const float w01 = (vy0 && vx1) ? msk * wy0 * wx1 : 0.0f;
            const float w10 = (vy1 && vx0) ? msk * wy1 * wx0 : 0.0f;
            const float w11 = (vy1 && vx1) ? msk * wy1 * wx1 : 0.0f;
            const _Float16* q00 = xT + xt_idx(b, cy0 + 1, cx0 + 1) + ch;
            const _Float16* q01 = xT + xt_idx(b, cy0 + 1, cx1 + 1) + ch;
            const _Float16* q10 = xT + xt_idx(b, cy1 + 1, cx0 + 1) + ch;
            const _Float16* q11 = xT + xt_idx(b, cy1 + 1, cx1 + 1) + ch;
            const v8h c00a = *(const v8h*)(q00), c00b = *(const v8h*)(q00 + 8);
            const v8h c01a = *(const v8h*)(q01), c01b = *(const v8h*)(q01 + 8);
            const v8h c10a = *(const v8h*)(q10), c10b = *(const v8h*)(q10 + 8);
            const v8h c11a = *(const v8h*)(q11), c11b = *(const v8h*)(q11 + 8);
            v8h ra, rb;
#pragma unroll
            for (int i = 0; i < 8; ++i) {
                float sa = w00 * (float)c00a[i] + w01 * (float)c01a[i]
                         + w10 * (float)c10a[i] + w11 * (float)c11a[i];
                float sb = w00 * (float)c00b[i] + w01 * (float)c01b[i]
                         + w10 * (float)c10b[i] + w11 * (float)c11b[i];
                ra[i] = (_Float16)sa;
                rb[i] = (_Float16)sb;
            }
            *(v8h*)&As[pm][ch]     = ra;
            *(v8h*)&As[pm][ch + 8] = rb;
        }
        __syncthreads();
        /* ---- WMMA: 8 K-blocks x (4 M-tiles x 2 N-tiles) per wave ---- */
#pragma unroll
        for (int cb = 0; cb < 8; ++cb) {
            const int kblock = kk * 8 + cb;
            if (cb + 1 < 8) {
                __builtin_prefetch(wB + (((size_t)(kblock + 1) * 16 + nt0) * 32 + lane) * 16, 0, 1);
            }
            const v16h bf0 = *(const v16h*)(wB + (((size_t)kblock * 16 + nt0) * 32 + lane) * 16);
            const v16h bf1 = *(const v16h*)(wB + (((size_t)kblock * 16 + nt1) * 32 + lane) * 16);
#pragma unroll
            for (int mt = 0; mt < 4; ++mt) {
                union { v16h v; v8h h[2]; } af;
                const _Float16* ap = &As[mt * 16 + m][cb * 32 + kb];
                af.h[0] = *(const v8h*)(ap);
                af.h[1] = *(const v8h*)(ap + 16);
                acc[mt][0] = wmma_f16(af.v, bf0, acc[mt][0]);
                acc[mt][1] = wmma_f16(af.v, bf1, acc[mt][1]);
            }
        }
        __syncthreads();   /* protect As before next tap overwrites it */
    }

    /* ---- epilogue: NCHW f32 store ---- */
    const int nlo = lane & 15;
    const int mhi = (lane >> 4) << 3;
    const int o0 = nt0 * 16 + nlo;
    const int o1 = nt1 * 16 + nlo;
#pragma unroll
    for (int mt = 0; mt < 4; ++mt) {
#pragma unroll
        for (int r = 0; r < 8; ++r) {
            const int xx = mt * 16 + r + mhi;     /* tile = 64 consecutive x in one row */
            out[(((size_t)b * CO_ + o0) * H_ + y) * W_ + xx] = acc[mt][0][r];
            out[(((size_t)b * CO_ + o1) * H_ + y) * W_ + xx] = acc[mt][1][r];
        }
    }
}

extern "C" void kernel_launch(void* const* d_in, const int* in_sizes, int n_in,
                              void* d_out, int out_size, void* d_ws, size_t ws_size,
                              hipStream_t stream) {
    (void)in_sizes; (void)n_in; (void)out_size; (void)ws_size;
    const float* x    = (const float*)d_in[0];
    const float* offw = (const float*)d_in[1];
    const float* offb = (const float*)d_in[2];
    const float* modw = (const float*)d_in[3];
    const float* modb = (const float*)d_in[4];
    const float* regw = (const float*)d_in[5];
    float* out = (float*)d_out;

    char* ws = (char*)d_ws;
    _Float16* xT  = (_Float16*)(ws + XT_OFF);
    _Float16* wB  = (_Float16*)(ws + WB_OFF);
    _Float16* wOM = (_Float16*)(ws + WOM_OFF);
    float*    om  = (float*)(ws + OM_OFF);

    k_halo   <<<B_ * HP_, 256, 0, stream>>>(xT);
    k_xpose  <<<B_ * H_ * 8, 256, 0, stream>>>(x, xT);
    k_packB  <<<(NKB * 16 * 32 * 16 + 255) / 256, 256, 0, stream>>>(regw, wB);
    k_packOM <<<(NKB * 2 * 32 * 16 + 255) / 256, 256, 0, stream>>>(offw, modw, wOM);
    k_offmask<<<NTILE_M / 4, 128, 0, stream>>>(xT, wOM, offb, modb, om);
    k_deform <<<NTILE_M / 4, 256, 0, stream>>>(xT, wB, om, out);
}
